// CrossAttentionMM_14688788152979
// MI455X (gfx1250) — matrix operand order repaired and checked by reading, not compiled, not run on hardware
//
#include <hip/hip_runtime.h>

// ---------------------------------------------------------------------------
// CDNA5 (gfx1250) cross-attention: bf16 WMMA pipeline + async global->LDS
//   q = x@Wq, k = ctx@Wk, v = ctx@Wv        (WMMA GEMM, f32 -> bf16 out)
//   flash attention per (b, head)           (WMMA QK^T + online softmax + WMMA PV)
//   out = attn@Wo + bo                      (WMMA GEMM, bf16 A async-staged, f32 out)
// ---------------------------------------------------------------------------

typedef __attribute__((ext_vector_type(16))) __bf16 v16bf;
typedef __attribute__((ext_vector_type(8)))  float  v8f;
typedef __attribute__((ext_vector_type(4)))  int    v4i;

union Frag { unsigned int u[8]; v16bf v; };

__device__ inline unsigned short f2bf(float f) {
  union { float f; unsigned int u; } x; x.f = f;
  unsigned int r = x.u + 0x7FFFu + ((x.u >> 16) & 1u);   // round-to-nearest-even
  return (unsigned short)(r >> 16);
}
__device__ inline unsigned int pack2(float a, float b) {
  return (unsigned int)f2bf(a) | ((unsigned int)f2bf(b) << 16);
}

// ---- CDNA5 async global->LDS copy (ASYNCcnt path), with safe fallback ------
#if defined(__has_builtin)
#if __has_builtin(__builtin_amdgcn_global_load_async_to_lds_b128) && \
    __has_builtin(__builtin_amdgcn_s_wait_asynccnt)
#define USE_ASYNC_LDS 1
#endif
#endif
#ifndef USE_ASYNC_LDS
#define USE_ASYNC_LDS 0
#endif

// int4 pointers in global (AS1) / LDS (AS3) address spaces, per builtin sig
typedef __attribute__((address_space(1))) v4i g_v4i;
typedef __attribute__((address_space(3))) v4i l_v4i;

__device__ inline void async_copy16(const void* gsrc, void* ldst) {
#if USE_ASYNC_LDS
  __builtin_amdgcn_global_load_async_to_lds_b128((g_v4i*)gsrc, (l_v4i*)ldst,
                                                 0, 0);
#else
  *(uint4*)ldst = *(const uint4*)gsrc;
#endif
}
__device__ inline void async_wait0() {
#if USE_ASYNC_LDS
  __builtin_amdgcn_s_wait_asynccnt(0);
#endif
}

// ---------------------------------------------------------------------------
// Generic WMMA GEMM: C[M,N] = A[M,K] * B[K,N] (+bias)
// Block: 256 threads (8 waves). Block tile 128x128, K-step 32.
// Wave grid 4(M) x 2(N); wave tile 32x64 -> 2x4 accumulators of 16x16 f32.
// A is f32 (converted while staging) or bf16 (async-copied to LDS);
// B is f32 (converted to bf16 in LDS); C is bf16 (u16) or f32.
// ---------------------------------------------------------------------------
template<bool A_IS_F32, bool C_IS_F32, bool HAS_BIAS>
__global__ __launch_bounds__(256)
void gemm_wmma_kernel(const void* __restrict__ A_, const float* __restrict__ Bm,
                      void* __restrict__ C_, const float* __restrict__ bias,
                      int M, int N, int K) {
  const int m0  = blockIdx.y * 128;
  const int n0  = blockIdx.x * 128;
  const int tid = threadIdx.x;
  const int lane = tid & 31;
  const int wid  = tid >> 5;
  const int h  = (lane >> 4) & 1;     // EXEC half
  const int ln = lane & 15;
  const int wm = (wid & 3) * 32;      // wave M offset in block tile
  const int wn = (wid >> 2) * 64;     // wave N offset in block tile

  // As rows padded to 80B (16B aligned for async B128, 20 mod 64 banks distinct)
  __shared__ __align__(16) unsigned int As[128][20]; // [row][K-pair]
  __shared__ unsigned int Bs[128][17];               // [col][K-pair] (transposed)

  const float* Af = (const float*)A_;

  v8f acc[2][4] = {};

  for (int kk = 0; kk < K; kk += 32) {
    // ---- stage A tile (128x32 bf16) ----
    if constexpr (A_IS_F32) {
      for (int idx = tid; idx < 2048; idx += 256) {
        int row = idx >> 4, kp = idx & 15;
        long o = (long)(m0 + row) * K + kk + 2 * kp;
        As[row][kp] = pack2(Af[o], Af[o + 1]);
      }
    } else {
      // bf16 source, identical layout: async DMA 16B chunks straight into LDS
      for (int idx = tid; idx < 512; idx += 256) {
        int row = idx >> 2, seg = idx & 3;
        const char* g =
            (const char*)A_ + ((long)(m0 + row) * K + kk) * 2 + seg * 16;
        async_copy16(g, (char*)&As[row][0] + seg * 16);
      }
    }
    // ---- stage B tile (32x128) transposed: Bs[n][kpair] ----
    for (int idx = tid; idx < 2048; idx += 256) {
      int n = idx & 127, kp = idx >> 7;
      float b0 = Bm[(long)(kk + 2 * kp)     * N + n0 + n];
      float b1 = Bm[(long)(kk + 2 * kp + 1) * N + n0 + n];
      Bs[n][kp] = pack2(b0, b1);
    }
    // prefetch next B tile rows into cache (global_prefetch_b8)
    if (kk + 32 < K)
      __builtin_prefetch(&Bm[(long)(kk + 32) * N + n0 + (tid & 127)], 0, 0);
    if constexpr (!A_IS_F32) async_wait0();
    __syncthreads();

    // ---- A fragments: 16x32 bf16, lane half picks K octet ----
    Frag af[2];
#pragma unroll
    for (int mi = 0; mi < 2; ++mi)
#pragma unroll
      for (int t = 0; t < 8; ++t)
        af[mi].u[t] = As[wm + mi * 16 + ln][(t >> 2) * 8 + h * 4 + (t & 3)];

#pragma unroll
    for (int ni = 0; ni < 4; ++ni) {
      Frag bf;
#pragma unroll
      for (int t = 0; t < 8; ++t)
        bf.u[t] = Bs[wn + ni * 16 + ln][(t >> 2) * 8 + h * 4 + (t & 3)];
#pragma unroll
      for (int mi = 0; mi < 2; ++mi)
        acc[mi][ni] = __builtin_amdgcn_wmma_f32_16x16x32_bf16(
            false, af[mi].v, false, bf.v, (short)0, acc[mi][ni], false, false);
    }
    __syncthreads();
  }

  // ---- epilogue: C/D layout row M = r + 8*h, col N = ln ----
#pragma unroll
  for (int mi = 0; mi < 2; ++mi)
#pragma unroll
    for (int ni = 0; ni < 4; ++ni)
#pragma unroll
      for (int r = 0; r < 8; ++r) {
        int row = m0 + wm + mi * 16 + r + 8 * h;
        int col = n0 + wn + ni * 16 + ln;
        float val = acc[mi][ni][r];
        if (HAS_BIAS) val += bias[col];
        if (C_IS_F32) ((float*)C_)[(long)row * N + col] = val;
        else ((unsigned short*)C_)[(long)row * N + col] = f2bf(val);
      }
}

// ---------------------------------------------------------------------------
// Flash attention: block = (b, head, 64 q-rows), 128 threads = 4 waves.
// Each wave owns a 16-row strip. KV walked in chunks of 64 with online softmax.
// Q and K tiles are byte-identical to their LDS layout -> async DMA staging.
// ---------------------------------------------------------------------------
__global__ __launch_bounds__(128)
void attn_wmma_kernel(const unsigned short* __restrict__ Q,
                      const unsigned short* __restrict__ Km,
                      const unsigned short* __restrict__ Vm,
                      unsigned short* __restrict__ O) {
  constexpr int SQ = 4096, SKV = 1024, INNER = 1024, DH = 64;
  const int qb   = blockIdx.x * 64;
  const int head = blockIdx.y;
  const int b    = blockIdx.z;
  const int tid  = threadIdx.x;
  const int lane = tid & 31, wid = tid >> 5;
  const int h = (lane >> 4) & 1, ln = lane & 15;

  // Qs/Ks rows padded to 144B (16B aligned, 36 mod 64 banks distinct)
  __shared__ __align__(16) unsigned int Qs[64][36];  // [q-row][d-pair]
  __shared__ __align__(16) unsigned int Ks[64][36];  // [kv-row][d-pair]
  __shared__ unsigned int Vs[64][33];                // [d][kv-pair] (transposed)
  __shared__ __align__(4) unsigned short Ps[4][16][64]; // per-wave P strip

  // ---- async-load Q strip once (64 rows x 128B head slice = 512 x 16B) ----
  for (int idx = tid; idx < 512; idx += 128) {
    int row = idx >> 3, seg = idx & 7;
    const char* g =
        (const char*)(Q + (long)(b * SQ + qb + row) * INNER + head * DH) +
        seg * 16;
    async_copy16(g, (char*)&Qs[row][0] + seg * 16);
  }
  async_wait0();
  __syncthreads();

  Frag qf[2];
#pragma unroll
  for (int ks = 0; ks < 2; ++ks)
#pragma unroll
    for (int t = 0; t < 8; ++t)
      qf[ks].u[t] = Qs[wid * 16 + ln][ks * 16 + (t >> 2) * 8 + h * 4 + (t & 3)];

  v8f o_acc[4] = {};
  float m_run[8], l_run[8];
#pragma unroll
  for (int r = 0; r < 8; ++r) { m_run[r] = -3.0e38f; l_run[r] = 0.0f; }

  for (int j = 0; j < SKV / 64; ++j) {
    const int kvb = j * 64;
    __syncthreads();  // previous chunk done reading Ks/Vs
    // K chunk: async DMA, layout identical to memory
    for (int idx = tid; idx < 512; idx += 128) {
      int row = idx >> 3, seg = idx & 7;
      const char* g =
          (const char*)(Km + (long)(b * SKV + kvb + row) * INNER + head * DH) +
          seg * 16;
      async_copy16(g, (char*)&Ks[row][0] + seg * 16);
    }
    // V chunk: transposed staging Vs[d][kv-pair]
    for (int idx = tid; idx < 64 * 32; idx += 128) {
      int d = idx & 63, kvp = idx >> 6;
      long o0 = (long)(b * SKV + kvb + 2 * kvp) * INNER + head * DH + d;
      Vs[d][kvp] = (unsigned int)Vm[o0] | ((unsigned int)Vm[o0 + INNER] << 16);
    }
    async_wait0();
    __syncthreads();

    // ---- S = Q K^T (16x64 strip per wave), K window = d in 2 steps of 32 ----
    v8f s[4] = {};
#pragma unroll
    for (int ks = 0; ks < 2; ++ks)
#pragma unroll
      for (int nt = 0; nt < 4; ++nt) {
        Frag kf;
#pragma unroll
        for (int t = 0; t < 8; ++t)
          kf.u[t] = Ks[nt * 16 + ln][ks * 16 + (t >> 2) * 8 + h * 4 + (t & 3)];
        s[nt] = __builtin_amdgcn_wmma_f32_16x16x32_bf16(
            false, qf[ks].v, false, kf.v, (short)0, s[nt], false, false);
      }

    // ---- online softmax: C-row r+8h spans 16 lanes of one half ----
#pragma unroll
    for (int r = 0; r < 8; ++r) {
      float mx = -3.0e38f;
#pragma unroll
      for (int nt = 0; nt < 4; ++nt) {
        float v = s[nt][r] * 0.125f;   // scale = DHEAD^-0.5
        s[nt][r] = v;
        mx = fmaxf(mx, v);
      }
#pragma unroll
      for (int off = 1; off < 16; off <<= 1) mx = fmaxf(mx, __shfl_xor(mx, off, 32));
      float mnew  = fmaxf(m_run[r], mx);
      float alpha = __expf(m_run[r] - mnew);
      m_run[r] = mnew;
      float rs = 0.0f;
#pragma unroll
      for (int nt = 0; nt < 4; ++nt) {
        float p = __expf(s[nt][r] - mnew);
        s[nt][r] = p;
        rs += p;
      }
#pragma unroll
      for (int off = 1; off < 16; off <<= 1) rs += __shfl_xor(rs, off, 32);
      l_run[r] = l_run[r] * alpha + rs;
#pragma unroll
      for (int dt = 0; dt < 4; ++dt) o_acc[dt][r] *= alpha;
#pragma unroll
      for (int nt = 0; nt < 4; ++nt)
        Ps[wid][r + 8 * h][nt * 16 + ln] = f2bf(s[nt][r]);
    }
    __syncthreads();  // publish Ps (cross-lane) before A-fragment reads

    // ---- O += P V : P is A (16 x 64 kv), V is B ([kv][d] -> Vs[d][kvpair]) ----
    const unsigned int* prow = (const unsigned int*)&Ps[wid][ln][0];
#pragma unroll
    for (int ks = 0; ks < 2; ++ks) {
      Frag pf;
#pragma unroll
      for (int t = 0; t < 8; ++t)
        pf.u[t] = prow[ks * 16 + (t >> 2) * 8 + h * 4 + (t & 3)];
#pragma unroll
      for (int dt = 0; dt < 4; ++dt) {
        Frag vf;
#pragma unroll
        for (int t = 0; t < 8; ++t)
          vf.u[t] = Vs[dt * 16 + ln][ks * 16 + (t >> 2) * 8 + h * 4 + (t & 3)];
        o_acc[dt] = __builtin_amdgcn_wmma_f32_16x16x32_bf16(
            false, pf.v, false, vf.v, (short)0, o_acc[dt], false, false);
      }
    }
  }

  // ---- normalize and store bf16 in [b, q, h*64+d] layout ----
#pragma unroll
  for (int r = 0; r < 8; ++r) {
    float inv = 1.0f / l_run[r];
    int row = qb + wid * 16 + r + 8 * h;
#pragma unroll
    for (int dt = 0; dt < 4; ++dt) {
      int d = dt * 16 + ln;
      O[(long)(b * SQ + row) * INNER + head * DH + d] = f2bf(o_acc[dt][r] * inv);
    }
  }
}

// ---------------------------------------------------------------------------
extern "C" void kernel_launch(void* const* d_in, const int* in_sizes, int n_in,
                              void* d_out, int out_size, void* d_ws, size_t ws_size,
                              hipStream_t stream) {
  (void)in_sizes; (void)n_in; (void)out_size; (void)ws_size;
  const float* x   = (const float*)d_in[0];
  const float* ctx = (const float*)d_in[1];
  const float* Wq  = (const float*)d_in[2];
  const float* Wk  = (const float*)d_in[3];
  const float* Wv  = (const float*)d_in[4];
  const float* Wo  = (const float*)d_in[5];
  const float* bo  = (const float*)d_in[6];
  float* out = (float*)d_out;

  constexpr int B = 4, SQ = 4096, SKV = 1024, QDIM = 1024, CDIM = 768,
                INNER = 1024, HEADS = 16;

  // workspace: q(32MB) k(8MB) v(8MB) attn_out(32MB), all bf16
  unsigned short* qb = (unsigned short*)d_ws;
  unsigned short* kb = qb + (size_t)B * SQ  * INNER;
  unsigned short* vb = kb + (size_t)B * SKV * INNER;
  unsigned short* ob = vb + (size_t)B * SKV * INNER;

  dim3 blk(256);
  // projections: A=f32, C=bf16, no bias
  gemm_wmma_kernel<true, false, false>
      <<<dim3(INNER / 128, (B * SQ) / 128), blk, 0, stream>>>(
          x, Wq, qb, nullptr, B * SQ, INNER, QDIM);
  gemm_wmma_kernel<true, false, false>
      <<<dim3(INNER / 128, (B * SKV) / 128), blk, 0, stream>>>(
          ctx, Wk, kb, nullptr, B * SKV, INNER, CDIM);
  gemm_wmma_kernel<true, false, false>
      <<<dim3(INNER / 128, (B * SKV) / 128), blk, 0, stream>>>(
          ctx, Wv, vb, nullptr, B * SKV, INNER, CDIM);

  attn_wmma_kernel<<<dim3(SQ / 64, HEADS, B), dim3(128), 0, stream>>>(qb, kb, vb, ob);

  // output projection: A=bf16 (async LDS staging), C=f32, +bias
  gemm_wmma_kernel<false, true, true>
      <<<dim3(QDIM / 128, (B * SQ) / 128), blk, 0, stream>>>(
          ob, Wo, out, bo, B * SQ, QDIM, INNER);
}
